// GQAttention_57724360458749
// MI455X (gfx1250) — matrix operand order repaired and checked by reading, not compiled, not run on hardware
//
#include <hip/hip_runtime.h>
#include <hip/hip_bf16.h>

// ---------------------------------------------------------------------------
// GQA attention block for MI455X (gfx1250, wave32, WMMA).
//   HIDDEN=2560, N_HEADS=32, N_KV=8, HD=80, B=2, S=2048
// Pipeline: cvt/transpose -> QKV GEMM (bf16 WMMA) -> RoPE+pack -> flash attn
//           (bf16 WMMA, online softmax) -> out GEMM (bf16 WMMA, f32 out).
// ---------------------------------------------------------------------------

#define HIDDEN 2560
#define NH     32
#define NKV    8
#define HD     80
#define HDP    96          // head dim padded to 3 * 32 for WMMA k-steps
#define BB     2
#define SS     2048
#define NTOK   (BB * SS)   // 4096
#define NQ     (NH * HD)   // 2560
#define NKVD   (NKV * HD)  // 640

typedef __bf16 bf16;
typedef __attribute__((ext_vector_type(16))) __bf16 bf16x16;
typedef __attribute__((ext_vector_type(8)))  float  f32x8;

struct u128 { unsigned int x, y, z, w; };        // POD 16-byte chunk
union FragBF { bf16x16 v; u128 u[2]; };          // one WMMA bf16 operand (32B)

// ---------------------------------------------------------------------------
// fp32 -> bf16 elementwise convert
// ---------------------------------------------------------------------------
__global__ void cvt_bf16_kernel(const float* __restrict__ in,
                                bf16* __restrict__ out, size_t n) {
  size_t i = (size_t)blockIdx.x * blockDim.x + threadIdx.x;
  if (i < n) out[i] = (bf16)in[i];
}

// Wt[n][k] = (bf16) W[k][n]   (W is [K][N] fp32)
__global__ void transpose_bf16_kernel(const float* __restrict__ W,
                                      bf16* __restrict__ Wt, int K, int N) {
  size_t total = (size_t)K * N;
  size_t idx = (size_t)blockIdx.x * blockDim.x + threadIdx.x;
  if (idx >= total) return;
  int k = (int)(idx % K);
  int n = (int)(idx / K);
  Wt[idx] = (bf16)W[(size_t)k * N + n];
}

// ---------------------------------------------------------------------------
// Generic bf16 WMMA GEMM:  C[M][N] = A[M][K] * Bt[N][K]^T
// block = 256 threads = 8 waves; wave w -> rows [m0, m0+16), cols [n0, n0+64)
// Fragment layouts follow CDNA5 ISA 7.12.2 (16-bit A 16x32, B 32x16).
// ---------------------------------------------------------------------------
template <bool OUT_F32>
__global__ __launch_bounds__(256) void gemm_bf16_kernel(
    const bf16* __restrict__ A, const bf16* __restrict__ Bt,
    void* __restrict__ C, int M, int N, int K) {
  const int lane = threadIdx.x & 31;
  const int wave = threadIdx.x >> 5;
  const int l16  = lane & 15;
  const int sub  = lane >> 4;                      // lane half selects K-half
  const int m0 = blockIdx.y * 128 + wave * 16;
  const int n0 = blockIdx.x * 64;

  const bf16* arow = A + (size_t)(m0 + l16) * K;   // A row for this lane
  f32x8 acc[4] = {};

  for (int k = 0; k < K; k += 32) {
    FragBF a;                                      // A frag: K {0..7,16..23} / {8..15,24..31}
    a.u[0] = *(const u128*)(arow + k + sub * 8);
    a.u[1] = *(const u128*)(arow + k + sub * 8 + 16);
    __builtin_prefetch(arow + k + 256, 0, 0);      // global_prefetch_b8
#pragma unroll
    for (int t = 0; t < 4; ++t) {
      const bf16* bcol = Bt + (size_t)(n0 + t * 16 + l16) * K + k + sub * 8;
      FragBF b;                                    // B frag: K {0..7,16..23} / {8..15,24..31}
      b.u[0] = *(const u128*)(bcol);
      b.u[1] = *(const u128*)(bcol + 16);
      acc[t] = __builtin_amdgcn_wmma_f32_16x16x32_bf16(
          false, a.v, false, b.v, (short)0, acc[t], false, false);
    }
  }

  // D layout: lanes 0-15 col=lane rows 0..7; lanes 16-31 col=lane-16 rows 8..15
  const int rowb = m0 + sub * 8;
#pragma unroll
  for (int t = 0; t < 4; ++t) {
    const int col = n0 + t * 16 + l16;
#pragma unroll
    for (int v = 0; v < 8; ++v) {
      size_t idx = (size_t)(rowb + v) * N + col;
      if constexpr (OUT_F32) ((float*)C)[idx] = acc[t][v];
      else                   ((bf16*)C)[idx]  = (bf16)acc[t][v];
    }
  }
}

// ---------------------------------------------------------------------------
// RoPE + repack into [b][head][s][HDP] bf16 (zero-padded d in [80,96))
// ---------------------------------------------------------------------------
__global__ void rope_pack_kernel(const bf16* __restrict__ raw,
                                 const float* __restrict__ cptr,
                                 const float* __restrict__ sptr,
                                 bf16* __restrict__ packed, int nheads) {
  size_t total = (size_t)BB * nheads * SS * HDP;
  size_t idx = (size_t)blockIdx.x * blockDim.x + threadIdx.x;
  if (idx >= total) return;
  int d = (int)(idx % HDP);
  size_t r = idx / HDP;
  int s = (int)(r % SS); r /= SS;
  int h = (int)(r % nheads);
  int b = (int)(r / nheads);
  bf16 outv = (bf16)0.0f;
  if (d < HD) {
    const bf16* row = raw + (size_t)(b * SS + s) * (nheads * HD) + h * HD;
    float x  = (float)row[d];
    float c  = cptr[s * HD + d];
    float sn = sptr[s * HD + d];
    float other = (d < HD / 2) ? -(float)row[d + HD / 2] : (float)row[d - HD / 2];
    outv = (bf16)(x * c + other * sn);
  }
  packed[idx] = outv;   // idx decomposes exactly as [b][h][s][HDP]
}

// Vt[b][kv][d][s] = Vraw[b*S+s][kv*HD+d]   (contiguous over keys for PV B-frags)
__global__ void v_transpose_kernel(const bf16* __restrict__ Vraw,
                                   bf16* __restrict__ Vt) {
  size_t total = (size_t)BB * NKV * HD * SS;
  size_t idx = (size_t)blockIdx.x * blockDim.x + threadIdx.x;
  if (idx >= total) return;
  int s = (int)(idx % SS);
  size_t r = idx / SS;
  int d  = (int)(r % HD); r /= HD;
  int kv = (int)(r % NKV);
  int b  = (int)(r / NKV);
  Vt[idx] = Vraw[(size_t)(b * SS + s) * NKVD + kv * HD + d];
}

// ---------------------------------------------------------------------------
// Flash attention: 1 wave per 16-row Q tile; online softmax; causal mask.
// QK^T: 3 WMMA k-steps over padded 96 head dim. PV: P staged through LDS
// (D-layout store -> A-layout ds_load_b128), 5 d-tiles of 16.
// ---------------------------------------------------------------------------
__global__ __launch_bounds__(128) void flash_attn_kernel(
    const bf16* __restrict__ Qp, const bf16* __restrict__ Kp,
    const bf16* __restrict__ Vt, bf16* __restrict__ Oo) {
  __shared__ alignas(16) bf16 lds_p[4][16 * 32];   // per-wave P tile (1 KB)

  const int lane = threadIdx.x & 31;
  const int wave = threadIdx.x >> 5;
  const int tile = blockIdx.x * 4 + wave;          // 8192 tiles total
  const int q0  = (tile & 127) * 16;               // S/16 = 128 q-tiles
  const int h   = (tile >> 7) & 31;
  const int b   = tile >> 12;
  const int kvh = h >> 2;                          // GQA group of 4
  const int l16 = lane & 15;
  const int sub = lane >> 4;

  // Q fragments (A operand), 3 k-chunks of 32 over padded head dim
  const bf16* qbase = Qp + ((size_t)((b * NH + h) * SS) + q0 + l16) * HDP;
  FragBF qf[3];
#pragma unroll
  for (int t = 0; t < 3; ++t) {
    qf[t].u[0] = *(const u128*)(qbase + t * 32 + sub * 8);
    qf[t].u[1] = *(const u128*)(qbase + t * 32 + sub * 8 + 16);
  }

  const bf16* kbase = Kp + (size_t)((b * NKV + kvh) * SS) * HDP;
  const bf16* vbase = Vt + (size_t)((b * NKV + kvh) * HD) * SS;
  bf16* myp = &lds_p[wave][0];

  f32x8 o0 = {}, o1 = {}, o2 = {}, o3 = {}, o4 = {};
  float m_s[8], l_s[8];
#pragma unroll
  for (int v = 0; v < 8; ++v) { m_s[v] = -1e30f; l_s[v] = 0.0f; }
  const float rscale = 0.11180339887498948f;       // 1/sqrt(80)

  for (int kb = 0; kb < q0 + 16; kb += 32) {       // 32 keys per iteration
    f32x8 s[2];
#pragma unroll
    for (int hh = 0; hh < 2; ++hh) {               // two 16-key S tiles
      f32x8 acc = {};
      const bf16* krow = kbase + (size_t)(kb + hh * 16 + l16) * HDP;
#pragma unroll
      for (int t = 0; t < 3; ++t) {                // B frag = K row (K^T col)
        FragBF kf;
        kf.u[0] = *(const u128*)(krow + t * 32 + sub * 8);
        kf.u[1] = *(const u128*)(krow + t * 32 + sub * 8 + 16);
        acc = __builtin_amdgcn_wmma_f32_16x16x32_bf16(
            false, qf[t].v, false, kf.v, (short)0, acc, false, false);
      }
      const int col = kb + hh * 16 + l16;          // key index of this lane
#pragma unroll
      for (int v = 0; v < 8; ++v) {                // scale + causal mask
        float val = acc[v] * rscale;
        const int row = q0 + sub * 8 + v;
        if (col > row) val = -1e9f;
        acc[v] = val;
      }
      s[hh] = acc;
    }

    // online softmax: row max / rescale / row sum (reduce over 16 key lanes;
    // xor masks < 16 keep the reduction within each lane half = row group)
    float alpha[8];
#pragma unroll
    for (int v = 0; v < 8; ++v) {
      float mx = fmaxf(s[0][v], s[1][v]);
#pragma unroll
      for (int off = 1; off < 16; off <<= 1)
        mx = fmaxf(mx, __shfl_xor(mx, off, 32));
      float mnew = fmaxf(m_s[v], mx);
      alpha[v] = __expf(m_s[v] - mnew);
      m_s[v] = mnew;
    }
#pragma unroll
    for (int v = 0; v < 8; ++v) {
      float p0 = __expf(s[0][v] - m_s[v]);
      float p1 = __expf(s[1][v] - m_s[v]);
      s[0][v] = p0; s[1][v] = p1;
      float rs = p0 + p1;
#pragma unroll
      for (int off = 1; off < 16; off <<= 1)
        rs += __shfl_xor(rs, off, 32);
      l_s[v] = l_s[v] * alpha[v] + rs;
      o0[v] *= alpha[v]; o1[v] *= alpha[v]; o2[v] *= alpha[v];
      o3[v] *= alpha[v]; o4[v] *= alpha[v];
    }

    // P: D-layout -> LDS -> A-layout fragment for the PV WMMA
#pragma unroll
    for (int hh = 0; hh < 2; ++hh)
#pragma unroll
      for (int v = 0; v < 8; ++v)
        myp[(sub * 8 + v) * 32 + hh * 16 + l16] = (bf16)s[hh][v];

    FragBF pf;
    const bf16* prow = myp + l16 * 32 + sub * 8;
    pf.u[0] = *(const u128*)(prow);                // ds_load_b128
    pf.u[1] = *(const u128*)(prow + 16);

    const bf16* vk = vbase + kb + sub * 8;
#pragma unroll
    for (int t = 0; t < 5; ++t) {                  // 5 d-tiles of 16 (HD=80)
      const bf16* vcol = vk + (size_t)(t * 16 + l16) * SS;
      FragBF vf;
      vf.u[0] = *(const u128*)(vcol);
      vf.u[1] = *(const u128*)(vcol + 16);
      f32x8& oa = (t == 0 ? o0 : t == 1 ? o1 : t == 2 ? o2 : t == 3 ? o3 : o4);
      oa = __builtin_amdgcn_wmma_f32_16x16x32_bf16(
          false, pf.v, false, vf.v, (short)0, oa, false, false);
    }
  }

  // normalize and store (row-major [token][h*80+d] bf16, feeds Wo GEMM)
  const size_t rowb = (size_t)(b * SS + q0 + sub * 8);
#pragma unroll
  for (int t = 0; t < 5; ++t) {
    const f32x8& oa = (t == 0 ? o0 : t == 1 ? o1 : t == 2 ? o2 : t == 3 ? o3 : o4);
#pragma unroll
    for (int v = 0; v < 8; ++v)
      Oo[(rowb + v) * NQ + h * HD + t * 16 + l16] = (bf16)(oa[v] / l_s[v]);
  }
}

// ---------------------------------------------------------------------------
// launcher
// ---------------------------------------------------------------------------
extern "C" void kernel_launch(void* const* d_in, const int* in_sizes, int n_in,
                              void* d_out, int out_size, void* d_ws, size_t ws_size,
                              hipStream_t stream) {
  (void)in_sizes; (void)n_in; (void)out_size; (void)ws_size;
  const float* x  = (const float*)d_in[0];
  const float* cf = (const float*)d_in[1];
  const float* sf = (const float*)d_in[2];
  const float* Wq = (const float*)d_in[3];
  const float* Wk = (const float*)d_in[4];
  const float* Wv = (const float*)d_in[5];
  const float* Wo = (const float*)d_in[6];
  float* out = (float*)d_out;

  char* ws = (char*)d_ws;
  size_t off = 0;
  auto alloc = [&](size_t elems) {
    bf16* p = (bf16*)(ws + off);
    off += ((elems * sizeof(bf16) + 255) & ~(size_t)255);
    return p;
  };
  bf16* xb   = alloc((size_t)NTOK * HIDDEN);
  bf16* WqT  = alloc((size_t)NQ * HIDDEN);
  bf16* WkT  = alloc((size_t)NKVD * HIDDEN);
  bf16* WvT  = alloc((size_t)NKVD * HIDDEN);
  bf16* WoT  = alloc((size_t)HIDDEN * NQ);
  bf16* Qraw = alloc((size_t)NTOK * NQ);
  bf16* Kraw = alloc((size_t)NTOK * NKVD);
  bf16* Vraw = alloc((size_t)NTOK * NKVD);
  bf16* Qp   = alloc((size_t)BB * NH * SS * HDP);
  bf16* Kp   = alloc((size_t)BB * NKV * SS * HDP);
  bf16* Vt   = alloc((size_t)BB * NKV * HD * SS);
  bf16* attnO = Qraw;   // Qraw is dead after rope_pack; reuse for attn output

  // 1. precision conversion / weight transposition (bf16, Bt[N][K] layout)
  { size_t n = (size_t)NTOK * HIDDEN;
    cvt_bf16_kernel<<<(unsigned)((n + 255) / 256), 256, 0, stream>>>(x, xb, n); }
  { size_t n = (size_t)HIDDEN * NQ;
    transpose_bf16_kernel<<<(unsigned)((n + 255) / 256), 256, 0, stream>>>(Wq, WqT, HIDDEN, NQ); }
  { size_t n = (size_t)HIDDEN * NKVD;
    transpose_bf16_kernel<<<(unsigned)((n + 255) / 256), 256, 0, stream>>>(Wk, WkT, HIDDEN, NKVD); }
  { size_t n = (size_t)HIDDEN * NKVD;
    transpose_bf16_kernel<<<(unsigned)((n + 255) / 256), 256, 0, stream>>>(Wv, WvT, HIDDEN, NKVD); }
  { size_t n = (size_t)NQ * HIDDEN;
    transpose_bf16_kernel<<<(unsigned)((n + 255) / 256), 256, 0, stream>>>(Wo, WoT, NQ, HIDDEN); }

  // 2. QKV projections (bf16 WMMA GEMMs)
  gemm_bf16_kernel<false><<<dim3(NQ / 64,   NTOK / 128), 256, 0, stream>>>(xb, WqT, Qraw, NTOK, NQ,   HIDDEN);
  gemm_bf16_kernel<false><<<dim3(NKVD / 64, NTOK / 128), 256, 0, stream>>>(xb, WkT, Kraw, NTOK, NKVD, HIDDEN);
  gemm_bf16_kernel<false><<<dim3(NKVD / 64, NTOK / 128), 256, 0, stream>>>(xb, WvT, Vraw, NTOK, NKVD, HIDDEN);

  // 3. RoPE + layout packing
  { size_t n = (size_t)BB * NH * SS * HDP;
    rope_pack_kernel<<<(unsigned)((n + 255) / 256), 256, 0, stream>>>(Qraw, cf, sf, Qp, NH); }
  { size_t n = (size_t)BB * NKV * SS * HDP;
    rope_pack_kernel<<<(unsigned)((n + 255) / 256), 256, 0, stream>>>(Kraw, cf, sf, Kp, NKV); }
  { size_t n = (size_t)BB * NKV * HD * SS;
    v_transpose_kernel<<<(unsigned)((n + 255) / 256), 256, 0, stream>>>(Vraw, Vt); }

  // 4. causal GQA flash attention (bf16 WMMA)
  flash_attn_kernel<<<dim3((BB * NH * (SS / 16)) / 4), 128, 0, stream>>>(Qp, Kp, Vt, attnO);

  // 5. output projection, fp32 result straight into d_out
  gemm_bf16_kernel<true><<<dim3(HIDDEN / 64, NTOK / 128), 256, 0, stream>>>(attnO, WoT, out, NTOK, HIDDEN, NQ);
}
